// MultiHeadAttention_27444841021700
// MI455X (gfx1250) — compile-verified, matrix operand order checked by reading, not run on hardware
//
#include <hip/hip_runtime.h>
#include <hip/hip_bf16.h>

// MHA forward for gfx1250 (MI455X): bf16 WMMA (v_wmma_f32_16x16x32_bf16) for all
// GEMM stages with 2x2 register blocking (32x32 per wave -> 16 FLOP/byte from
// cache), f32 accumulate, full att matrix materialized in d_out per reference.
//
// Shapes: B=2, S=2048, D=1024, H=16, HD=64.

#define Bq   2
#define Sq   2048
#define Dq   1024
#define Hq   16
#define HDq  64

typedef __attribute__((ext_vector_type(16))) __bf16 v16bf;
typedef __attribute__((ext_vector_type(8)))  float  v8f;

struct __attribute__((aligned(16))) U128 { unsigned int x0, x1, x2, x3; };

#define WMMA_BF16(A, B, C) \
    __builtin_amdgcn_wmma_f32_16x16x32_bf16(false, (A), false, (B), (short)0, (C), false, false)

// ---------------------------------------------------------------------------
// Fragment loaders.  CDNA5 wave32 16-bit A/B fragment layout (CDNA5 ISA):
//   lanes 0-15 : row = lane,    K-chunks {k+0..7,  k+16..23}
//   lanes 16-31: row = lane-16, K-chunks {k+8..15, k+24..31}
// Both A (row-major M x K) and B (B[kk][n] = src[n][kk], src row-major N x K)
// use the same addressing: "row" is M for A, N for B.
// ---------------------------------------------------------------------------
__device__ __forceinline__ v16bf load_frag_bf16(const __bf16* base, int row,
                                                int ld, int kbase, int lane) {
    int r     = lane & 15;
    int khalf = (lane >> 4) << 3;                 // 0 or 8
    const __bf16* p = base + (size_t)(row + r) * ld + kbase + khalf;
    union { v16bf v; U128 u[2]; } f;
    f.u[0] = *reinterpret_cast<const U128*>(p);       // K chunk +0 (or +8)
    f.u[1] = *reinterpret_cast<const U128*>(p + 16);  // K chunk +16 (or +24)
    return f.v;
}

// A-fragment from an f32 matrix (att) with on-the-fly f32->bf16 conversion.
__device__ __forceinline__ v16bf load_frag_f32cvt(const float* base, int row,
                                                  int ld, int kbase, int lane) {
    int r     = lane & 15;
    int khalf = (lane >> 4) << 3;
    const float* p = base + (size_t)(row + r) * ld + kbase + khalf;
    v16bf v;
#pragma unroll
    for (int i = 0; i < 8; ++i) v[i] = (__bf16)p[i];
#pragma unroll
    for (int i = 0; i < 8; ++i) v[8 + i] = (__bf16)p[16 + i];
    return v;
}

// ---------------------------------------------------------------------------
// Stage 0: f32 -> bf16 conversion (one pass over x, W_attn, W_proj)
// ---------------------------------------------------------------------------
__global__ void cvt_f32_bf16(const float* __restrict__ in,
                             __bf16* __restrict__ out, int n) {
    int i = blockIdx.x * blockDim.x + threadIdx.x;
    int stride = gridDim.x * blockDim.x;
    for (; i < n; i += stride) out[i] = (__bf16)in[i];
}

// ---------------------------------------------------------------------------
// Stage 1: QKV GEMM  qkv = x @ W_attn^T   [4096,1024] x [1024,3072]
// 2x2 register blocking: each wave owns a 32x32 output block.
// Writes q,k head-major [B,H,S,HD] and v transposed [B,H,HD,S], all bf16.
// ---------------------------------------------------------------------------
__device__ __forceinline__ void qkv_store_tile(const v8f& c, int m0, int n0, int lane,
                                               __bf16* __restrict__ q,
                                               __bf16* __restrict__ k,
                                               __bf16* __restrict__ vt) {
    int n    = n0 + (lane & 15);
    int sec  = n >> 10;              // 0:q 1:k 2:v (uniform per tile)
    int dcol = n & (Dq - 1);
    int h  = dcol >> 6;
    int hd = dcol & (HDq - 1);
#pragma unroll
    for (int j = 0; j < 8; ++j) {
        int m  = m0 + j + ((lane >> 4) << 3);
        int b_ = m >> 11;
        int s  = m & (Sq - 1);
        __bf16 val = (__bf16)c[j];
        size_t bh = (size_t)(b_ * Hq + h);
        if (sec == 0)      q [bh * Sq * HDq + (size_t)s * HDq + hd] = val;
        else if (sec == 1) k [bh * Sq * HDq + (size_t)s * HDq + hd] = val;
        else               vt[bh * HDq * Sq + (size_t)hd * Sq + s]  = val;
    }
}

__global__ void qkv_gemm(const __bf16* __restrict__ xb,
                         const __bf16* __restrict__ wab,
                         __bf16* __restrict__ q, __bf16* __restrict__ k,
                         __bf16* __restrict__ vt) {
    const int NB = (3 * Dq) / 32;                       // 96 N blocks
    int blk  = blockIdx.x * 8 + (threadIdx.x >> 5);
    int lane = threadIdx.x & 31;
    int m0 = (blk / NB) << 5;
    int n0 = (blk % NB) << 5;

    v8f c00 = {}, c01 = {}, c10 = {}, c11 = {};
    for (int kk = 0; kk < Dq; kk += 32) {
        v16bf a0 = load_frag_bf16(xb,  m0,      Dq, kk, lane);
        v16bf a1 = load_frag_bf16(xb,  m0 + 16, Dq, kk, lane);
        v16bf b0 = load_frag_bf16(wab, n0,      Dq, kk, lane);
        v16bf b1 = load_frag_bf16(wab, n0 + 16, Dq, kk, lane);
        c00 = WMMA_BF16(a0, b0, c00);
        c01 = WMMA_BF16(a0, b1, c01);
        c10 = WMMA_BF16(a1, b0, c10);
        c11 = WMMA_BF16(a1, b1, c11);
    }
    qkv_store_tile(c00, m0,      n0,      lane, q, k, vt);
    qkv_store_tile(c01, m0,      n0 + 16, lane, q, k, vt);
    qkv_store_tile(c10, m0 + 16, n0,      lane, q, k, vt);
    qkv_store_tile(c11, m0 + 16, n0 + 16, lane, q, k, vt);
}

// ---------------------------------------------------------------------------
// Stage 2: scores = (q @ k^T) * 1/sqrt(HD), f32 into d_out att region.
// 32x32 output per wave; K=64 -> 8 WMMAs with full fragment reuse.
// ---------------------------------------------------------------------------
__device__ __forceinline__ void store_f32_tile(const v8f& c, float* __restrict__ dst,
                                               int ld, int m0, int n0, int lane,
                                               float scale) {
    int n = n0 + (lane & 15);
#pragma unroll
    for (int j = 0; j < 8; ++j) {
        int m = m0 + j + ((lane >> 4) << 3);
        dst[(size_t)m * ld + n] = c[j] * scale;
    }
}

__global__ void qk_scores(const __bf16* __restrict__ q,
                          const __bf16* __restrict__ k,
                          float* __restrict__ att) {
    const int BLKS_PER_BH = (Sq / 32) * (Sq / 32);      // 64 * 64 = 4096
    int blk  = blockIdx.x * 8 + (threadIdx.x >> 5);
    int lane = threadIdx.x & 31;
    int bh  = blk / BLKS_PER_BH;
    int rem = blk % BLKS_PER_BH;
    int m0 = (rem >> 6) << 5;
    int n0 = (rem & 63) << 5;

    const __bf16* qb = q + (size_t)bh * Sq * HDq;
    const __bf16* kb = k + (size_t)bh * Sq * HDq;
    float* ab = att + (size_t)bh * Sq * Sq;

    v8f c00 = {}, c01 = {}, c10 = {}, c11 = {};
#pragma unroll
    for (int kk = 0; kk < HDq; kk += 32) {
        v16bf a0 = load_frag_bf16(qb, m0,      HDq, kk, lane);
        v16bf a1 = load_frag_bf16(qb, m0 + 16, HDq, kk, lane);
        v16bf b0 = load_frag_bf16(kb, n0,      HDq, kk, lane);
        v16bf b1 = load_frag_bf16(kb, n0 + 16, HDq, kk, lane);
        c00 = WMMA_BF16(a0, b0, c00);
        c01 = WMMA_BF16(a0, b1, c01);
        c10 = WMMA_BF16(a1, b0, c10);
        c11 = WMMA_BF16(a1, b1, c11);
    }
    const float scale = 0.125f;                          // 1/sqrt(64)
    store_f32_tile(c00, ab, Sq, m0,      n0,      lane, scale);
    store_f32_tile(c01, ab, Sq, m0,      n0 + 16, lane, scale);
    store_f32_tile(c10, ab, Sq, m0 + 16, n0,      lane, scale);
    store_f32_tile(c11, ab, Sq, m0 + 16, n0 + 16, lane, scale);
}

// ---------------------------------------------------------------------------
// Stage 3: row softmax over att (in place, f32). One 256-thread block per row.
// ---------------------------------------------------------------------------
__global__ void softmax_rows(float* __restrict__ att) {
    __shared__ float red[256];
    float* p = att + (size_t)blockIdx.x * Sq;
    int t = threadIdx.x;
    float vals[8];
    float m = -INFINITY;
#pragma unroll
    for (int i = 0; i < 8; ++i) { vals[i] = p[t + 256 * i]; m = fmaxf(m, vals[i]); }
    red[t] = m; __syncthreads();
    for (int s = 128; s > 0; s >>= 1) {
        if (t < s) red[t] = fmaxf(red[t], red[t + s]);
        __syncthreads();
    }
    float rowmax = red[0]; __syncthreads();
    float sum = 0.f;
#pragma unroll
    for (int i = 0; i < 8; ++i) { vals[i] = __expf(vals[i] - rowmax); sum += vals[i]; }
    red[t] = sum; __syncthreads();
    for (int s = 128; s > 0; s >>= 1) {
        if (t < s) red[t] += red[t + s];
        __syncthreads();
    }
    float inv = 1.0f / red[0];
#pragma unroll
    for (int i = 0; i < 8; ++i) p[t + 256 * i] = vals[i] * inv;
}

// ---------------------------------------------------------------------------
// Stage 4: y_ctx = att @ v.  A from f32 att (cvt in regs), B from vt rows.
// 32x32 output per wave, K loop = 2048/32 = 64 steps x 4 WMMAs.
// Output bf16 [B,S,D] layout.
// ---------------------------------------------------------------------------
__global__ void att_v(const float* __restrict__ att,
                      const __bf16* __restrict__ vt,
                      __bf16* __restrict__ yctx) {
    const int BLKS_PER_BH = (Sq / 32) * (HDq / 32);      // 64 * 2 = 128
    int blk  = blockIdx.x * 8 + (threadIdx.x >> 5);
    int lane = threadIdx.x & 31;
    int bh  = blk / BLKS_PER_BH;
    int rem = blk % BLKS_PER_BH;
    int m0 = (rem >> 1) << 5;
    int n0 = (rem & 1) << 5;

    const float*  ab = att + (size_t)bh * Sq * Sq;
    const __bf16* vb = vt  + (size_t)bh * HDq * Sq;

    v8f c00 = {}, c01 = {}, c10 = {}, c11 = {};
    for (int kk = 0; kk < Sq; kk += 32) {
        v16bf a0 = load_frag_f32cvt(ab, m0,      Sq, kk, lane);
        v16bf a1 = load_frag_f32cvt(ab, m0 + 16, Sq, kk, lane);
        v16bf b0 = load_frag_bf16 (vb, n0,      Sq, kk, lane);
        v16bf b1 = load_frag_bf16 (vb, n0 + 16, Sq, kk, lane);
        c00 = WMMA_BF16(a0, b0, c00);
        c01 = WMMA_BF16(a0, b1, c01);
        c10 = WMMA_BF16(a1, b0, c10);
        c11 = WMMA_BF16(a1, b1, c11);
    }

    int b_ = bh >> 4, h = bh & 15;
#pragma unroll
    for (int t = 0; t < 4; ++t) {
        const v8f& c = (t == 0) ? c00 : (t == 1) ? c01 : (t == 2) ? c10 : c11;
        int mo = m0 + ((t >> 1) << 4);
        int n  = h * HDq + n0 + ((t & 1) << 4) + (lane & 15);
#pragma unroll
        for (int j = 0; j < 8; ++j) {
            int s = mo + j + ((lane >> 4) << 3);
            yctx[((size_t)(b_ * Sq + s)) * Dq + n] = (__bf16)c[j];
        }
    }
}

// ---------------------------------------------------------------------------
// Stage 5: y = y_ctx @ W_proj^T + b_proj, f32 into d_out y region.
// 32x32 output per wave.
// ---------------------------------------------------------------------------
__global__ void proj_gemm(const __bf16* __restrict__ yctx,
                          const __bf16* __restrict__ wpb,
                          const float* __restrict__ bias,
                          float* __restrict__ y) {
    const int NB = Dq / 32;                              // 32
    int blk  = blockIdx.x * 8 + (threadIdx.x >> 5);
    int lane = threadIdx.x & 31;
    int m0 = (blk / NB) << 5;
    int n0 = (blk % NB) << 5;

    v8f c00 = {}, c01 = {}, c10 = {}, c11 = {};
    for (int kk = 0; kk < Dq; kk += 32) {
        v16bf a0 = load_frag_bf16(yctx, m0,      Dq, kk, lane);
        v16bf a1 = load_frag_bf16(yctx, m0 + 16, Dq, kk, lane);
        v16bf b0 = load_frag_bf16(wpb,  n0,      Dq, kk, lane);
        v16bf b1 = load_frag_bf16(wpb,  n0 + 16, Dq, kk, lane);
        c00 = WMMA_BF16(a0, b0, c00);
        c01 = WMMA_BF16(a0, b1, c01);
        c10 = WMMA_BF16(a1, b0, c10);
        c11 = WMMA_BF16(a1, b1, c11);
    }

#pragma unroll
    for (int t = 0; t < 4; ++t) {
        const v8f& c = (t == 0) ? c00 : (t == 1) ? c01 : (t == 2) ? c10 : c11;
        int mo = m0 + ((t >> 1) << 4);
        int n  = n0 + ((t & 1) << 4) + (lane & 15);
        float bn = bias[n];
#pragma unroll
        for (int j = 0; j < 8; ++j) {
            int m = mo + j + ((lane >> 4) << 3);
            y[(size_t)m * Dq + n] = c[j] + bn;
        }
    }
}

// ---------------------------------------------------------------------------
extern "C" void kernel_launch(void* const* d_in, const int* in_sizes, int n_in,
                              void* d_out, int out_size, void* d_ws, size_t ws_size,
                              hipStream_t stream) {
    const float* x      = (const float*)d_in[0];   // [B,S,D]      4,194,304
    const float* W_attn = (const float*)d_in[1];   // [3D,D]       3,145,728
    const float* W_proj = (const float*)d_in[2];   // [D,D]        1,048,576
    const float* b_proj = (const float*)d_in[3];   // [D]

    float* y   = (float*)d_out;                    // [B,S,D]
    float* att = y + (size_t)Bq * Sq * Dq;         // [B,H,S,S]

    // Workspace carve-up (bf16), total 48 MB.
    __bf16* xb   = (__bf16*)d_ws;
    __bf16* wab  = xb   + (size_t)Bq * Sq * Dq;        // +4,194,304
    __bf16* wpb  = wab  + (size_t)3 * Dq * Dq;         // +3,145,728
    __bf16* q    = wpb  + (size_t)Dq * Dq;             // +1,048,576
    __bf16* k    = q    + (size_t)Bq * Hq * Sq * HDq;  // +4,194,304
    __bf16* vt   = k    + (size_t)Bq * Hq * Sq * HDq;  // +4,194,304
    __bf16* yctx = vt   + (size_t)Bq * Hq * Sq * HDq;  // +4,194,304

    // Stage 0: convert inputs to bf16 once.
    cvt_f32_bf16<<<2048, 256, 0, stream>>>(x,      xb,  Bq * Sq * Dq);
    cvt_f32_bf16<<<2048, 256, 0, stream>>>(W_attn, wab, 3 * Dq * Dq);
    cvt_f32_bf16<<<1024, 256, 0, stream>>>(W_proj, wpb, Dq * Dq);

    // Stage 1: QKV GEMM. 128x96 blocks of 32x32 = 12288 waves.
    qkv_gemm<<<1536, 256, 0, stream>>>(xb, wab, q, k, vt);

    // Stage 2: QK^T scores. 32 bh * 4096 blocks = 131072 waves.
    qk_scores<<<16384, 256, 0, stream>>>(q, k, att);

    // Stage 3: softmax over B*H*S = 65536 rows.
    softmax_rows<<<65536, 256, 0, stream>>>(att);

    // Stage 4: att @ V. 32 bh * 128 blocks = 4096 waves.
    att_v<<<512, 256, 0, stream>>>(att, vt, yctx);

    // Stage 5: output projection. 128x32 blocks = 4096 waves.
    proj_gemm<<<512, 256, 0, stream>>>(yctx, wpb, b_proj, y);
}
